// MambaBlock_27006754357575
// MI455X (gfx1250) — matrix operand order repaired and checked by reading, not compiled, not run on hardware
//
#include <hip/hip_runtime.h>
#include <math.h>

// ---------------------------------------------------------------------------
// MambaBlock for MI455X (gfx1250, wave32, WMMA bf16 + async LDS staging).
// Shapes: B=1, L=2048, d_model=1024, d_inner=2048, d_state=16, d_conv=4.
//
// Pipeline:
//   0) convert x, in_proj_w, dt_proj_w, out_proj_w to bf16 (one pass each)
//   1) xz  = x @ in_proj_w^T                  (WMMA)         -> fp32
//   2) xc  = silu(causal_dwconv(xz[:,:2048]) + conv_b)       -> fp32 + bf16
//   3) dt  = softplus(xc @ dt_proj_w^T + dt_proj_b) (WMMA)   -> fp32
//   4) per-row fused SSM -> y (bf16)
//   5) out = y @ out_proj_w^T                 (WMMA)         -> fp32
//
// GEMM stages A/B tiles (128x32 bf16 each) into LDS with double buffering via
// GLOBAL_LOAD_ASYNC_TO_LDS_B128 (ASYNCcnt), feeding v_wmma from ds_load_b128.
// ---------------------------------------------------------------------------

typedef __attribute__((ext_vector_type(16))) __bf16          v16bf;
typedef __attribute__((ext_vector_type(8)))  float           v8f;
typedef __attribute__((ext_vector_type(16))) unsigned short  v16u;
typedef __attribute__((ext_vector_type(8)))  unsigned short  v8u;
typedef __attribute__((ext_vector_type(4)))  unsigned short  v4u;
typedef __attribute__((ext_vector_type(4)))  int             v4i;
typedef unsigned short ushort_t;

// pointer-to-int4 in global (AS1) / LDS (AS3) address spaces
typedef __attribute__((address_space(1))) v4i* gv4i_p;
typedef __attribute__((address_space(3))) v4i* lv4i_p;

#if defined(__has_builtin)
#if __has_builtin(__builtin_amdgcn_global_load_async_to_lds_b128)
#define ASYNC_LDS 1
#else
#define ASYNC_LDS 0
#endif
#else
#define ASYNC_LDS 0
#endif

#if ASYNC_LDS
template <int N> __device__ __forceinline__ void wait_asynccnt() {
#if __has_builtin(__builtin_amdgcn_s_wait_asynccnt)
    __builtin_amdgcn_s_wait_asynccnt(N);
#else
    asm volatile("s_wait_asynccnt %0" ::"n"(N) : "memory");
#endif
}
#endif

__device__ __forceinline__ unsigned short f2bf(float f) {
    union { float f; unsigned int u; } c; c.f = f;
    unsigned int u = c.u;
    u += 0x7fffu + ((u >> 16) & 1u);            // round-to-nearest-even
    return (unsigned short)(u >> 16);
}

__device__ __forceinline__ v16bf pack_frag(v8u lo, v8u hi) {
    v16u u = __builtin_shufflevector(lo, hi, 0, 1, 2, 3, 4, 5, 6, 7,
                                     8, 9, 10, 11, 12, 13, 14, 15);
    return __builtin_bit_cast(v16bf, u);
}

// C = A(M,K) * W(N,K)^T in bf16 with fp32 accum.
// ACT == 0: plain fp32 store.  ACT == 1: softplus(v + bias[n]).
// Block: 256 threads = 8 waves (2 in M x 4 in N); block tile 128x128;
// per-wave 64x32 = 4x2 WMMA tiles; K stepped by 32.
// A/B 128x32 bf16 tiles staged in LDS (double buffered, async copies).
template <int ACT>
__global__ __launch_bounds__(256) void gemm_bf16_wmma(
    const ushort_t* __restrict__ A, const ushort_t* __restrict__ Bw,
    const float* __restrict__ bias, float* __restrict__ C,
    int M, int N, int K)
{
    __shared__ ushort_t ldsA[2][128 * 32];   // [row][k] row-major, 8KB each
    __shared__ ushort_t ldsB[2][128 * 32];

    const int tid  = threadIdx.x;
    const int wave = tid >> 5;
    const int lane = tid & 31;
    const int half = lane >> 4;
    const int lm   = lane & 15;

    const int mblk = blockIdx.y * 128;
    const int nblk = blockIdx.x * 128;
    const int wm   = (wave >> 2) * 64;       // wave M offset inside tile
    const int wn   = (wave & 3) * 32;        // wave N offset inside tile

    // tile-copy mapping: thread -> (row, 16-elem column pair)
    const int crow = tid >> 1;               // 0..127
    const int ccol = (tid & 1) * 16;         // 0 or 16

    auto issue_copy = [&](int buf, int k0) {
        const ushort_t* gA = A + (size_t)(mblk + crow) * K + k0 + ccol;
        const ushort_t* gB = Bw + (size_t)(nblk + crow) * K + k0 + ccol;
        ushort_t* lA = &ldsA[buf][crow * 32 + ccol];
        ushort_t* lB = &ldsB[buf][crow * 32 + ccol];
#if ASYNC_LDS
        __builtin_amdgcn_global_load_async_to_lds_b128(
            (gv4i_p)(gA),     (lv4i_p)(lA),     0, 0);
        __builtin_amdgcn_global_load_async_to_lds_b128(
            (gv4i_p)(gA + 8), (lv4i_p)(lA + 8), 0, 0);
        __builtin_amdgcn_global_load_async_to_lds_b128(
            (gv4i_p)(gB),     (lv4i_p)(lB),     0, 0);
        __builtin_amdgcn_global_load_async_to_lds_b128(
            (gv4i_p)(gB + 8), (lv4i_p)(lB + 8), 0, 0);
#else
        v8u a0 = *(const v8u*)(gA);
        v8u a1 = *(const v8u*)(gA + 8);
        v8u b0 = *(const v8u*)(gB);
        v8u b1 = *(const v8u*)(gB + 8);
        *(v8u*)(lA)     = a0;
        *(v8u*)(lA + 8) = a1;
        *(v8u*)(lB)     = b0;
        *(v8u*)(lB + 8) = b1;
#endif
    };

    v8f acc[4][2];
    const v8f vzero = {0.f, 0.f, 0.f, 0.f, 0.f, 0.f, 0.f, 0.f};
#pragma unroll
    for (int i = 0; i < 4; ++i)
#pragma unroll
        for (int j = 0; j < 2; ++j) acc[i][j] = vzero;

    const int nsteps = K >> 5;
    issue_copy(0, 0);

    for (int s = 0; s < nsteps; ++s) {
        const int  cur  = s & 1;
        const bool more = (s + 1) < nsteps;
        if (more) issue_copy(cur ^ 1, (s + 1) << 5);
#if ASYNC_LDS
        if (more) wait_asynccnt<4>();        // prev batch done, next in flight
        else      wait_asynccnt<0>();
#endif
        __syncthreads();

        const ushort_t* tA = ldsA[cur];
        const ushort_t* tB = ldsB[cur];
        v16bf af[4], bf[2];
#pragma unroll
        for (int i = 0; i < 4; ++i) {
            const int r = wm + i * 16 + lm;
            // A frag: lane holds K = {half*8+0..7} and {16+half*8+0..7}
            v8u lo = *(const v8u*)(tA + r * 32 + half * 8);
            v8u hi = *(const v8u*)(tA + r * 32 + 16 + half * 8);
            af[i] = pack_frag(lo, hi);
        }
#pragma unroll
        for (int j = 0; j < 2; ++j) {
            const int r = wn + j * 16 + lm;
            // B frag: lane holds K = {half*8+0..7} and {16+half*8+0..7}
            v8u lo = *(const v8u*)(tB + r * 32 + half * 8);
            v8u hi = *(const v8u*)(tB + r * 32 + 16 + half * 8);
            bf[j] = pack_frag(lo, hi);
        }
#pragma unroll
        for (int i = 0; i < 4; ++i)
#pragma unroll
            for (int j = 0; j < 2; ++j)
                acc[i][j] = __builtin_amdgcn_wmma_f32_16x16x32_bf16(
                    false, af[i], false, bf[j], (short)0, acc[i][j],
                    false, false);
        __syncthreads();   // protect cur buffer from next issue
    }

#pragma unroll
    for (int i = 0; i < 4; ++i) {
#pragma unroll
        for (int j = 0; j < 2; ++j) {
            const int n = nblk + wn + j * 16 + lm;
            float bv = 0.f;
            if (ACT == 1) bv = bias[n];
#pragma unroll
            for (int r = 0; r < 8; ++r) {
                const int m = mblk + wm + i * 16 + half * 8 + r;
                float v = acc[i][j][r];
                if (ACT == 1) {
                    v += bv;
                    v = (v > 20.f) ? v : log1pf(expf(v));   // softplus
                }
                C[(size_t)m * N + n] = v;
            }
        }
    }
}

// One-pass fp32 -> bf16 (RNE), 4 elements / thread.
__global__ __launch_bounds__(256) void f32_to_bf16_kernel(
    const float* __restrict__ src, ushort_t* __restrict__ dst, int n4)
{
    const int i = blockIdx.x * 256 + threadIdx.x;
    if (i >= n4) return;
    const float4 v = ((const float4*)src)[i];
    v4u o;
    o[0] = f2bf(v.x); o[1] = f2bf(v.y); o[2] = f2bf(v.z); o[3] = f2bf(v.w);
    ((v4u*)dst)[i] = o;
}

// Causal depthwise conv (d_conv=4) + bias + silu over xz[:, :2048].
__global__ __launch_bounds__(256) void conv_silu_kernel(
    const float* __restrict__ xz, const float* __restrict__ cw,
    const float* __restrict__ cb, float* __restrict__ xc,
    ushort_t* __restrict__ xc_bf)
{
    const int idx = blockIdx.x * 256 + threadIdx.x;   // over L*2048
    const int l = idx >> 11;
    const int d = idx & 2047;
    float acc = cb[d];
#pragma unroll
    for (int t = 0; t < 4; ++t) {
        const int ll = l - 3 + t;
        if (ll >= 0) acc += xz[(size_t)ll * 4096 + d] * cw[d * 4 + t];
    }
    const float s = acc / (1.f + expf(-acc));         // silu
    xc[(size_t)idx]    = s;
    xc_bf[(size_t)idx] = f2bf(s);
}

// One block per row l:
//   s = sum_d xc*dt; bc[n] = sum_d xc * x_proj_w[n][d]  (n=0..31)
//   coef = s * sum_n bc[n]*bc[n+16]
//   y_bf[l][d] = bf16( (coef + xc[l][d]*D[d]) * silu(z[l][d]) )
__global__ __launch_bounds__(256) void ssm_fuse_kernel(
    const float* __restrict__ xz, const float* __restrict__ xc,
    const float* __restrict__ dt, const float* __restrict__ xpw,
    const float* __restrict__ Dp, ushort_t* __restrict__ y_bf)
{
    const int l    = blockIdx.x;
    const int tid  = threadIdx.x;
    const int wave = tid >> 5;
    const int lane = tid & 31;

    const float* xcrow = xc + (size_t)l * 2048;
    const float* dtrow = dt + (size_t)l * 2048;

    float sp = 0.f;
    float bca[32];
#pragma unroll
    for (int n = 0; n < 32; ++n) bca[n] = 0.f;

    for (int j = tid; j < 2048; j += 256) {
        const float xv = xcrow[j];
        const float dv = dtrow[j];
        sp += xv * dv;
#pragma unroll
        for (int n = 0; n < 32; ++n) bca[n] += xv * xpw[n * 2048 + j];
    }

#pragma unroll
    for (int off = 16; off > 0; off >>= 1) {
        sp += __shfl_xor(sp, off, 32);
#pragma unroll
        for (int n = 0; n < 32; ++n) bca[n] += __shfl_xor(bca[n], off, 32);
    }

    __shared__ float s_w[8];
    __shared__ float bc_w[8][32];
    __shared__ float bcf[32];
    __shared__ float coef_s;

    if (lane == 0) {
        s_w[wave] = sp;
#pragma unroll
        for (int n = 0; n < 32; ++n) bc_w[wave][n] = bca[n];
    }
    __syncthreads();

    if (tid < 32) {
        float a = 0.f;
#pragma unroll
        for (int w = 0; w < 8; ++w) a += bc_w[w][tid];
        bcf[tid] = a;
    }
    __syncthreads();

    if (tid == 0) {
        float s = 0.f;
#pragma unroll
        for (int w = 0; w < 8; ++w) s += s_w[w];
        float dot = 0.f;
#pragma unroll
        for (int n = 0; n < 16; ++n) dot += bcf[n] * bcf[n + 16];
        coef_s = s * dot;
    }
    __syncthreads();

    const float coef = coef_s;
    for (int j = tid; j < 2048; j += 256) {
        const float xv = xcrow[j];
        const float zv = xz[(size_t)l * 4096 + 2048 + j];
        const float sz = zv / (1.f + expf(-zv));      // silu(z)
        y_bf[(size_t)l * 2048 + j] = f2bf((coef + xv * Dp[j]) * sz);
    }
}

extern "C" void kernel_launch(void* const* d_in, const int* in_sizes, int n_in,
                              void* d_out, int out_size, void* d_ws, size_t ws_size,
                              hipStream_t stream) {
    (void)in_sizes; (void)n_in; (void)out_size; (void)ws_size;

    const float* x          = (const float*)d_in[0];  // (L, 1024)
    const float* in_proj_w  = (const float*)d_in[1];  // (4096, 1024)
    const float* conv_w     = (const float*)d_in[2];  // (2048, 1, 4)
    const float* conv_b     = (const float*)d_in[3];  // (2048,)
    const float* x_proj_w   = (const float*)d_in[4];  // (32, 2048)
    const float* dt_proj_w  = (const float*)d_in[5];  // (2048, 2048)
    const float* dt_proj_b  = (const float*)d_in[6];  // (2048,)
    const float* out_proj_w = (const float*)d_in[7];  // (1024, 2048)
    /* d_in[8] = A_log: dead code in tustin mode */
    const float* D_param    = (const float*)d_in[9];  // (2048,)

    float* out = (float*)d_out;                       // (L, 1024)

    const int L = 2048, DM = 1024, DI = 2048;

    // fp32 scratch
    float* fws = (float*)d_ws;
    float* xz  = fws;                                 // L*4096
    float* xc  = xz + (size_t)L * (2 * DI);           // L*2048
    float* dt  = xc + (size_t)L * DI;                 // L*2048
    // bf16 scratch
    ushort_t* ub       = (ushort_t*)(dt + (size_t)L * DI);
    ushort_t* x_bf     = ub;                          // L*1024
    ushort_t* w_in_bf  = x_bf    + (size_t)L * DM;    // 4096*1024
    ushort_t* w_dt_bf  = w_in_bf + (size_t)(2 * DI) * DM;  // 2048*2048
    ushort_t* w_out_bf = w_dt_bf + (size_t)DI * DI;   // 1024*2048
    ushort_t* xc_bf    = w_out_bf + (size_t)DM * DI;  // L*2048
    ushort_t* y_bf     = xc_bf   + (size_t)L * DI;    // L*2048

    dim3 blk(256);

    // 0) one-pass bf16 conversions
    {
        const int nx  = L * DM / 4;
        const int nwi = (2 * DI) * DM / 4;
        const int nwd = DI * DI / 4;
        const int nwo = DM * DI / 4;
        f32_to_bf16_kernel<<<(nx  + 255) / 256, blk, 0, stream>>>(x,          x_bf,     nx);
        f32_to_bf16_kernel<<<(nwi + 255) / 256, blk, 0, stream>>>(in_proj_w,  w_in_bf,  nwi);
        f32_to_bf16_kernel<<<(nwd + 255) / 256, blk, 0, stream>>>(dt_proj_w,  w_dt_bf,  nwd);
        f32_to_bf16_kernel<<<(nwo + 255) / 256, blk, 0, stream>>>(out_proj_w, w_out_bf, nwo);
    }

    // 1) xz = x @ in_proj_w^T         (M=2048, N=4096, K=1024)
    gemm_bf16_wmma<0><<<dim3((2 * DI) / 128, L / 128), blk, 0, stream>>>(
        x_bf, w_in_bf, nullptr, xz, L, 2 * DI, DM);

    // 2) xc = silu(dwconv(xz[:, :DI]) + conv_b)  (fp32 + bf16)
    conv_silu_kernel<<<(L * DI) / 256, blk, 0, stream>>>(xz, conv_w, conv_b,
                                                         xc, xc_bf);

    // 3) dt = softplus(xc @ dt_proj_w^T + dt_proj_b)  (M=2048, N=2048, K=2048)
    gemm_bf16_wmma<1><<<dim3(DI / 128, L / 128), blk, 0, stream>>>(
        xc_bf, w_dt_bf, dt_proj_b, dt, L, DI, DI);

    // 4) fused SSM per row -> y_bf
    ssm_fuse_kernel<<<L, blk, 0, stream>>>(xz, xc, dt, x_proj_w, D_param, y_bf);

    // 5) out = y @ out_proj_w^T        (M=2048, N=1024, K=2048)
    gemm_bf16_wmma<0><<<dim3(DM / 128, L / 128), blk, 0, stream>>>(
        y_bf, w_out_bf, nullptr, out, L, DM, DI);
}
